// Conv2d_int8_EST_9354438770983
// MI455X (gfx1250) — 8-bit matrix operand order repaired and checked by reading, compiled, not run on hardware
//
#include <hip/hip_runtime.h>
#include <stdint.h>


typedef __attribute__((ext_vector_type(8))) int v8i;

#define CIN      64
#define COUT     64
#define HW       14
#define BATCH    8
#define KSZ      3
#define KDIM     (CIN * KSZ * KSZ)
#define SPATIAL  (HW * HW)
#define NTOT     (BATCH * SPATIAL)
#define NT_M     (COUT / 16)
#define NT_N     (NTOT / 16)
#define KSTEPS   (KDIM / 64)

#define AMAX_X_OFF 0
#define AMAX_W_OFF 4
#define QW_OFF     64
#define QP_OFF     (QW_OFF + COUT * KDIM)

__global__ void ws_init_kernel(unsigned* amax) {
    if (threadIdx.x < 2) amax[threadIdx.x] = 0u;
}

__global__ void absmax_kernel(const float* __restrict__ p, int n, unsigned* out) {
    unsigned m = 0u;
    for (int i = blockIdx.x * blockDim.x + threadIdx.x; i < n;
         i += gridDim.x * blockDim.x) {
        unsigned v = __float_as_uint(fabsf(p[i]));
        m = (v > m) ? v : m;
    }
    #pragma unroll
    for (int off = 16; off > 0; off >>= 1) {
        unsigned o = (unsigned)__shfl_xor((int)m, off, 32);
        m = (o > m) ? o : m;
    }
    if ((threadIdx.x & 31) == 0) atomicMax(out, m);
}

__device__ __forceinline__ signed char quant1(float v, float inv) {
    float q = rintf(v * inv);
    q = fminf(fmaxf(q, -128.f), 127.f);
    return (signed char)(int)q;
}

__global__ void quant_w_kernel(const float* __restrict__ w,
                               const unsigned* __restrict__ amax,
                               signed char* __restrict__ qw, int n) {
    float m   = __uint_as_float(amax[1]);
    float inv = (m > 0.f) ? 127.0f / m : 0.f;
    int i = blockIdx.x * blockDim.x + threadIdx.x;
    if (i < n) qw[i] = quant1(w[i], inv);
}

__global__ void quant_patch_kernel(const float* __restrict__ x,
                                   const unsigned* __restrict__ amax,
                                   signed char* __restrict__ qp) {
    float m   = __uint_as_float(amax[0]);
    float inv = (m > 0.f) ? 127.0f / m : 0.f;
    int i = blockIdx.x * blockDim.x + threadIdx.x;
    if (i >= NTOT * KDIM) return;
    int n  = i / KDIM;
    int k  = i - n * KDIM;
    int b  = n / SPATIAL;
    int s  = n - b * SPATIAL;
    int oh = s / HW, ow = s - oh * HW;
    int c  = k / 9;
    int r  = k - c * 9;
    int kh = r / 3, kw = r - kh * 3;
    int ih = oh + kh - 1, iw = ow + kw - 1;
    float v = 0.f;
    if ((unsigned)ih < (unsigned)HW && (unsigned)iw < (unsigned)HW)
        v = x[((b * CIN + c) * HW + ih) * HW + iw];
    qp[i] = quant1(v, inv);
}

__global__ __launch_bounds__(32)
void conv_gemm_wmma_kernel(const signed char* __restrict__ qw,
                           const signed char* __restrict__ qp,
                           const unsigned* __restrict__ amax,
                           const float* __restrict__ bias,
                           float* __restrict__ out) {
    const int tile  = blockIdx.x;
    const int mtile = tile % NT_M;
    const int ntile = tile / NT_M;
    const int lane  = threadIdx.x;
    const int l16   = lane & 15;
    const int h     = lane >> 4;

    const int mrow = mtile * 16 + l16;
    const int ncol = ntile * 16 + l16;
    const int* __restrict__ arow = (const int*)(qw + mrow * KDIM);
    const int* __restrict__ bcol = (const int*)(qp + ncol * KDIM);

    v8i acc = {0, 0, 0, 0, 0, 0, 0, 0};

    #pragma unroll
    for (int ks = 0; ks < KSTEPS; ++ks) {
        const int kb = ks * 64;
        v8i A, B;
        #pragma unroll
        for (int v = 0; v < 8; ++v) {
            int ka = kb + ((v >> 1) << 4) + (h << 3) + ((v & 1) << 2);
            A[v] = arow[ka >> 2];
        }
        #pragma unroll
        for (int v = 0; v < 8; ++v) {
            int kbb = kb + ((v >> 1) << 4) + (h << 3) + ((v & 1) << 2);
            B[v] = bcol[kbb >> 2];
        }
        acc = __builtin_amdgcn_wmma_i32_16x16x64_iu8(true, A, true, B, acc,
                                                     false, false);
    }

    const float sx = __uint_as_float(amax[0]) * (1.0f / 127.0f);
    const float sw = __uint_as_float(amax[1]) * (1.0f / 127.0f);
    const float sc = sx * sw;

    const int b  = ncol / SPATIAL;
    const int sp = ncol - b * SPATIAL;
    #pragma unroll
    for (int r = 0; r < 8; ++r) {
        const int o = mtile * 16 + r + (h << 3);
        out[(b * COUT + o) * SPATIAL + sp] = sc * (float)acc[r] + bias[o];
    }
}

extern "C" void kernel_launch(void* const* d_in, const int* in_sizes, int n_in,
                              void* d_out, int out_size, void* d_ws, size_t ws_size,
                              hipStream_t stream) {
    const float* x    = (const float*)d_in[0];
    const float* w    = (const float*)d_in[1];
    const float* bias = (const float*)d_in[2];

    unsigned*    amax = (unsigned*)d_ws;
    signed char* qw   = (signed char*)d_ws + QW_OFF;
    signed char* qp   = (signed char*)d_ws + QP_OFF;
    float*       out  = (float*)d_out;

    const int nx = BATCH * CIN * HW * HW;
    const int nw = COUT * KDIM;
    const int np = NTOT * KDIM;

    ws_init_kernel<<<1, 64, 0, stream>>>(amax);
    absmax_kernel<<<128, 256, 0, stream>>>(x, nx, amax + 0);
    absmax_kernel<<<48, 256, 0, stream>>>(w, nw, amax + 1);
    quant_w_kernel<<<(nw + 255) / 256, 256, 0, stream>>>(w, amax, qw, nw);
    quant_patch_kernel<<<(np + 255) / 256, 256, 0, stream>>>(x, amax, qp);
    conv_gemm_wmma_kernel<<<NT_M * NT_N, 32, 0, stream>>>(qw, qp, amax, bias, out);
}
